// ICNNObsDecoder_80006650790345
// MI455X (gfx1250) — matrix operand order repaired and checked by reading, not compiled, not run on hardware
//
#include <hip/hip_runtime.h>

typedef __attribute__((ext_vector_type(16))) _Float16 v16h;
typedef __attribute__((ext_vector_type(8)))  _Float16 v8h;
typedef __attribute__((ext_vector_type(8)))  float    v8f;

// ---------------- problem constants ----------------
#define BB     8192
#define STATE  64
#define OBS    64
#define DD     128     // STATE+OBS
#define HID    384
#define NSTEP  8
#define ADAM_B1 0.9f
#define ADAM_B2 0.999f
#define ADAM_LR 5e-4f
#define ADAM_EPS 1e-8f

#define ROWS   16      // rows per workgroup

#define LOG2E_F 1.4426950408889634f
#define LN2_F   0.6931471805599453f

// ---------------- workspace layout (offsets in halfs) ----------------
static constexpr size_t OWZ0F = 0;                 // K=128 N=384 : 4*24*512  = 49152
static constexpr size_t OWZ1F = 49152;             // K=384 N=384 : 12*24*512 = 147456
static constexpr size_t OWZ2F = 196608;
static constexpr size_t OWZ3F = 344064;
static constexpr size_t OWX0F = 491520;            // K=128 N=384
static constexpr size_t OWX1F = 540672;
static constexpr size_t OWX2F = 589824;
static constexpr size_t OWZ3B = 638976;            // K=384 N=384 (transposed use)
static constexpr size_t OWZ2B = 786432;
static constexpr size_t OWZ1B = 933888;
static constexpr size_t OWZ0B = 1081344;           // K=384 N=128 : 12*8*512 = 49152
static constexpr size_t OWX2B = 1130496;
static constexpr size_t OWX1B = 1179648;
static constexpr size_t OWX0B = 1228800;
static constexpr size_t OHALF_END = 1277952;       // halfs
static constexpr size_t OF32B = OHALF_END * 2;     // byte offset of f32 vec region
// f32 region: [0..383] = sq(wz_w4), [384..511] = sq(wx_w3)

// ---------------- LDS layout (bytes), ROWS=16 ----------------
static constexpr unsigned L_XS  = 0;        // half [16][128]   4096
static constexpr unsigned L_Z0  = 4096;     // half [16][384]  12288
static constexpr unsigned L_Z1  = 16384;
static constexpr unsigned L_Z2  = 28672;
static constexpr unsigned L_Z3  = 40960;
static constexpr unsigned L_DA  = 53248;    // half [16][384]
static constexpr unsigned L_DB  = 65536;
static constexpr unsigned L_GX  = 77824;    // float [16][64]   4096 (also energy scratch)
static constexpr unsigned L_YO  = 81920;    // float [16][64]
static constexpr unsigned L_MO  = 86016;
static constexpr unsigned L_VO  = 90112;
static constexpr unsigned L_BY  = 94208;
static constexpr unsigned L_EE  = 98304;    // float [16]
static constexpr unsigned L_BE  = 98368;
static constexpr unsigned L_FL  = 98432;    // int [16]
static constexpr unsigned LDS_BYTES = 98496;

// ---------------- prep kernels ----------------
// Pack squared weights into WMMA B-fragment layout:
//   out[((kt*NT + nt)*32 + lane)*16 + j]  holds element (k = kt*32 + (lane>>4)*8 + (j<8 ? j : j+8),
//                                                        n = nt*16 + (lane&15))
__global__ void icnn_pack_w(const float* __restrict__ W, _Float16* __restrict__ out,
                            int NT, int Wcols, int transp, int total) {
    int idx = blockIdx.x * blockDim.x + threadIdx.x;
    if (idx >= total) return;
    int j  = idx & 15;
    int l  = (idx >> 4) & 31;
    int f  = idx >> 9;
    int nt = f % NT, kt = f / NT;
    int n  = nt * 16 + (l & 15);
    int k  = kt * 32 + ((l >> 4) << 3) + (j < 8 ? j : j + 8);
    float w = transp ? W[(size_t)k * Wcols + n] : W[(size_t)n * Wcols + k];
    out[idx] = (_Float16)(w * w);
}

__global__ void icnn_sq_vec(const float* __restrict__ w4, const float* __restrict__ wx3,
                            float* __restrict__ out) {
    int i = blockIdx.x * blockDim.x + threadIdx.x;
    if (i < HID)            out[i] = w4[i] * w4[i];
    else if (i < HID + DD) { int j = i - HID; out[i] = wx3[j] * wx3[j]; }
}

// ---------------- device helpers ----------------
// Opaque-offset: blocks LICM from hoisting weight loads across the step loop
// while PRESERVING pointer provenance (so loads lower to global_load, not flat).
__device__ __forceinline__ const _Float16* opq(const _Float16* p) {
    size_t z = 0;
    asm volatile("" : "+s"(z));
    return p + z;
}

// Raw-hardware softplus: max(p,0) + ln(1+exp(-|p|)); arg of log in [1,2] so the
// bare v_exp_f32/v_log_f32 ops are accurate — no denorm-fix blob.
__device__ __forceinline__ float softplus_f(float p) {
    float e = __builtin_amdgcn_exp2f(-__builtin_fabsf(p) * LOG2E_F);
    return fmaxf(p, 0.f) + __builtin_amdgcn_logf(1.f + e) * LN2_F;
}
// sigmoid(pre) given z = softplus(pre):  s = 1 - exp(-z)
__device__ __forceinline__ float sig_from_z(float z) {
    return 1.f - __builtin_amdgcn_exp2f(-z * LOG2E_F);
}

// Load a 16x32 f16 A-fragment from LDS (row-major, `stride` halfs per row).
// lane<16 -> M=lane, K = {0..7, 16..23}; lane>=16 -> M=lane-16, K={8..15, 24..31}
__device__ __forceinline__ v16h load_A_lds(const _Float16* base, int stride,
                                           int k0, int lane) {
    int m  = lane & 15;
    int kb = k0 + ((lane >> 4) << 3);
    const _Float16* p = base + m * stride + kb;
    v8h lo = *(const v8h*)p;         // ds_load_b128
    v8h hi = *(const v8h*)(p + 16);
    v16h a;
#pragma unroll
    for (int i = 0; i < 8; ++i) { a[i] = lo[i]; a[i + 8] = hi[i]; }
    return a;
}

// Three-tile accumulate: one A load feeds 3 WMMAs (tiles nt0, nt0+1, nt0+2).
// B loads hit consecutive 1KiB-apart fragments off one scalar base.
__device__ __forceinline__ void mm3(v8f& c0, v8f& c1, v8f& c2,
                                    const _Float16* A, int strideA,
                                    const _Float16* Bf, int NT, int nt0, int KT, int lane) {
    Bf = opq(Bf);
#pragma unroll 1
    for (int kt = 0; kt < KT; ++kt) {
        v16h a = load_A_lds(A, strideA, kt * 32, lane);
        const _Float16* bp = Bf + (((size_t)kt * NT + nt0) * 32 + (size_t)lane) * 16;
        v16h b0 = *(const v16h*)bp;
        v16h b1 = *(const v16h*)(bp + 512);
        v16h b2 = *(const v16h*)(bp + 1024);
        c0 = __builtin_amdgcn_wmma_f32_16x16x32_f16(false, a, false, b0, (short)0, c0, false, false);
        c1 = __builtin_amdgcn_wmma_f32_16x16x32_f16(false, a, false, b1, (short)0, c1, false, false);
        c2 = __builtin_amdgcn_wmma_f32_16x16x32_f16(false, a, false, b2, (short)0, c2, false, false);
    }
}

// Single-tile accumulate (for the obs-gradient tile).
__device__ __forceinline__ void mm1(v8f& c, const _Float16* A, int strideA,
                                    const _Float16* Bf, int NT, int nt, int KT, int lane) {
    Bf = opq(Bf);
#pragma unroll 1
    for (int kt = 0; kt < KT; ++kt) {
        v16h a = load_A_lds(A, strideA, kt * 32, lane);
        v16h b = *(const v16h*)(Bf + (((size_t)kt * NT + nt) * 32 + (size_t)lane) * 16);
        c = __builtin_amdgcn_wmma_f32_16x16x32_f16(false, a, false, b, (short)0, c, false, false);
    }
}

// pre = z_prev @ Wz^T + x @ Wx^T + bias ; z_out = softplus(pre); 3 tiles/wave
__device__ __forceinline__ void fwd_layer(const _Float16* Az, int KTz, const _Float16* Bz,
                                          const _Float16* xs, const _Float16* Bx, int KTx,
                                          const float* __restrict__ bias,
                                          _Float16* zout, int q, int lane) {
    v8f c0 = {}, c1 = {}, c2 = {};
    if (Bz) mm3(c0, c1, c2, Az, HID, Bz, 24, q * 3, KTz, lane);
    mm3(c0, c1, c2, xs, DD, Bx, 24, q * 3, KTx, lane);
    int nc = lane & 15, mh = (lane >> 4) << 3;
#pragma unroll
    for (int i = 0; i < 3; ++i) {
        v8f& c = (i == 0) ? c0 : (i == 1) ? c1 : c2;
        int n = (q * 3 + i) * 16 + nc;
        float b = bias[n];
#pragma unroll
        for (int r = 0; r < 8; ++r)
            zout[(r + mh) * HID + n] = (_Float16)softplus_f(c[r] + b);
    }
}

// d_out = (d_in @ Wz_bwd) * sigmoid(pre_prev); 3 tiles/wave
__device__ __forceinline__ void bwd_dlayer(const _Float16* dIn, const _Float16* Bb,
                                           const _Float16* zprev, _Float16* dOut,
                                           int q, int lane) {
    v8f c0 = {}, c1 = {}, c2 = {};
    mm3(c0, c1, c2, dIn, HID, Bb, 24, q * 3, 12, lane);
    int nc = lane & 15, mh = (lane >> 4) << 3;
#pragma unroll
    for (int i = 0; i < 3; ++i) {
        v8f& c = (i == 0) ? c0 : (i == 1) ? c1 : c2;
        int n = (q * 3 + i) * 16 + nc;
#pragma unroll
        for (int r = 0; r < 8; ++r) {
            int m = r + mh;
            float s = sig_from_z((float)zprev[m * HID + n]);
            dOut[m * HID + n] = (_Float16)(c[r] * s);
        }
    }
}

// ---------------- main kernel ----------------
__global__ __launch_bounds__(256)
void icnn_main(const float* __restrict__ state,
               const float* __restrict__ bz0,
               const float* __restrict__ bx0,
               const float* __restrict__ bx1,
               const float* __restrict__ bx2,
               const _Float16* __restrict__ wsh,
               const float* __restrict__ fvec,
               float* __restrict__ out) {
    extern __shared__ char smem[];
    _Float16* xs = (_Float16*)(smem + L_XS);
    _Float16* z0 = (_Float16*)(smem + L_Z0);
    _Float16* z1 = (_Float16*)(smem + L_Z1);
    _Float16* z2 = (_Float16*)(smem + L_Z2);
    _Float16* z3 = (_Float16*)(smem + L_Z3);
    _Float16* dA = (_Float16*)(smem + L_DA);
    _Float16* dB = (_Float16*)(smem + L_DB);
    float* gx  = (float*)(smem + L_GX);
    float* yo  = (float*)(smem + L_YO);
    float* mo  = (float*)(smem + L_MO);
    float* vo  = (float*)(smem + L_VO);
    float* by  = (float*)(smem + L_BY);
    float* ee  = (float*)(smem + L_EE);
    float* bee = (float*)(smem + L_BE);
    int*   flg = (int*)  (smem + L_FL);

    const int tid  = threadIdx.x;
    const int lane = tid & 31;
    const int q    = tid >> 5;    // wave id: owns N-tiles 3q..3q+2 of 24
    const int wg   = blockIdx.x;

    // ---- init: x = [state, 0], y=m=v=0, best=0, best_e=inf ----
    for (int e = tid; e < ROWS * OBS; e += 256) {
        int r = e >> 6, j = e & 63;
        float sv = state[((size_t)wg * ROWS + r) * STATE + j];
        xs[r * DD + j]         = (_Float16)sv;
        xs[r * DD + STATE + j] = (_Float16)0.f;
        yo[e] = 0.f; mo[e] = 0.f; vo[e] = 0.f; by[e] = 0.f;
    }
    if (tid < ROWS) bee[tid] = __builtin_inff();
    __syncthreads();

    float b1t = 1.f, b2t = 1.f;
#pragma unroll 1
    for (int t = 0; t <= NSTEP; ++t) {
        // Opaque integer offset: defeats cross-step LICM of weight loads while
        // keeping global address-space provenance.
        size_t woff = 0;
        asm volatile("" : "+s"(woff));
        const _Float16* wp = wsh + woff;

        // ---------- forward ----------
        fwd_layer(nullptr, 0, nullptr, xs, wp + OWZ0F, 4, bz0, z0, q, lane);       __syncthreads();
        fwd_layer(z0, 12, wp + OWZ1F, xs, wp + OWX0F, 4, bx0, z1, q, lane);        __syncthreads();
        fwd_layer(z1, 12, wp + OWZ2F, xs, wp + OWX1F, 4, bx1, z2, q, lane);        __syncthreads();
        fwd_layer(z2, 12, wp + OWZ3F, xs, wp + OWX2F, 4, bx2, z3, q, lane);        __syncthreads();

        // ---------- energy (deterministic two-stage reduction) ----------
        {   // 16 threads per row x 16 rows; partials stashed in gx scratch
            int r = tid >> 4, part = tid & 15;
            float s = 0.f;
#pragma unroll 1
            for (int i = 0; i < 24; ++i) { int k = part * 24 + i; s += (float)z3[r * HID + k] * fvec[k]; }
#pragma unroll 1
            for (int i = 0; i < 8;  ++i) { int k = part * 8 + i;  s += (float)xs[r * DD + k] * fvec[HID + k]; }
            gx[r * 16 + part] = s;
        }
        __syncthreads();
        if (tid < ROWS) {
            float s = 0.f;
#pragma unroll
            for (int i = 0; i < 16; ++i) s += gx[tid * 16 + i];
            ee[tid] = s;
            int better = s < bee[tid];
            if (better) bee[tid] = s;
            flg[tid] = better;
        }
        __syncthreads();
        for (int e = tid; e < ROWS * OBS; e += 256)
            if (flg[e >> 6]) by[e] = yo[e];
        __syncthreads();

        if (t == NSTEP) break;

        // ---------- backward ----------
        b1t *= ADAM_B1; b2t *= ADAM_B2;
        // d3 = sq(wz_w4) * sigmoid(pre3);  16 threads/row, 24 cols each (no div/mod)
        {
            int r = tid >> 4, nb = (tid & 15) * 24;
#pragma unroll 1
            for (int i = 0; i < 24; ++i) {
                int n = nb + i;
                float z = (float)z3[r * HID + n];
                dA[r * HID + n] = (_Float16)(fvec[n] * sig_from_z(z));
            }
        }
        __syncthreads();

        v8f gxC = {};
        bwd_dlayer(dA, wp + OWZ3B, z2, dB, q, lane);                     // d2 <- d3
        if (q < 4) mm1(gxC, dA, HID, wp + OWX2B, 8, 4 + q, 12, lane);    // gx += d3 @ Wx2
        __syncthreads();
        bwd_dlayer(dB, wp + OWZ2B, z1, dA, q, lane);                     // d1 <- d2
        if (q < 4) mm1(gxC, dB, HID, wp + OWX1B, 8, 4 + q, 12, lane);    // gx += d2 @ Wx1
        __syncthreads();
        bwd_dlayer(dA, wp + OWZ1B, z0, dB, q, lane);                     // d0 <- d1
        if (q < 4) mm1(gxC, dA, HID, wp + OWX0B, 8, 4 + q, 12, lane);    // gx += d1 @ Wx0
        __syncthreads();
        if (q < 4) {
            mm1(gxC, dB, HID, wp + OWZ0B, 8, 4 + q, 12, lane);           // gx += d0 @ Wz0
            int n = (4 + q) * 16 + (lane & 15);   // obs column, 64..127
            int j = n - STATE;
#pragma unroll
            for (int r = 0; r < 8; ++r) {
                int m = r + ((lane >> 4) << 3);
                gx[m * OBS + j] = gxC[r] + fvec[HID + n];   // + sq(wx_w3)
            }
        }
        __syncthreads();

        // ---------- Adam update on observables ----------
        float ic1 = 1.f / (1.f - b1t), ic2 = 1.f / (1.f - b2t);
        for (int e = tid; e < ROWS * OBS; e += 256) {
            int r = e >> 6, j = e & 63;
            float g  = gx[e];
            float m_ = mo[e] * ADAM_B1 + (1.f - ADAM_B1) * g;
            float v_ = vo[e] * ADAM_B2 + (1.f - ADAM_B2) * g * g;
            mo[e] = m_; vo[e] = v_;
            float y = yo[e] - ADAM_LR * (m_ * ic1) / (sqrtf(v_ * ic2) + ADAM_EPS);
            yo[e] = y;
            xs[r * DD + STATE + j] = (_Float16)y;
        }
        __syncthreads();
    }

    // ---------- output best observables ----------
    for (int e = tid; e < ROWS * OBS; e += 256)
        out[((size_t)wg * ROWS + (e >> 6)) * OBS + (e & 63)] = by[e];
}

// ---------------- host launcher ----------------
extern "C" void kernel_launch(void* const* d_in, const int* in_sizes, int n_in,
                              void* d_out, int out_size, void* d_ws, size_t ws_size,
                              hipStream_t stream) {
    const float* state = (const float*)d_in[0];
    const float* wz_w0 = (const float*)d_in[1];
    const float* wz_b0 = (const float*)d_in[2];
    const float* wz_w1 = (const float*)d_in[3];
    const float* wz_w2 = (const float*)d_in[4];
    const float* wz_w3 = (const float*)d_in[5];
    const float* wz_w4 = (const float*)d_in[6];
    const float* wx_w0 = (const float*)d_in[7];
    const float* wx_b0 = (const float*)d_in[8];
    const float* wx_w1 = (const float*)d_in[9];
    const float* wx_b1 = (const float*)d_in[10];
    const float* wx_w2 = (const float*)d_in[11];
    const float* wx_b2 = (const float*)d_in[12];
    const float* wx_w3 = (const float*)d_in[13];

    _Float16* wsh = (_Float16*)d_ws;
    float* fvec = (float*)((char*)d_ws + OF32B);

    auto pack = [&](const float* W, size_t off, int K, int N, int Wcols, int transp) {
        int total = (K / 32) * (N / 16) * 512;
        icnn_pack_w<<<(total + 255) / 256, 256, 0, stream>>>(W, wsh + off, N / 16, Wcols, transp, total);
    };
    // forward orientation (K = input dim, N = output dim, element = Wsq[n][k])
    pack(wz_w0, OWZ0F, DD,  HID, DD,  0);
    pack(wz_w1, OWZ1F, HID, HID, HID, 0);
    pack(wz_w2, OWZ2F, HID, HID, HID, 0);
    pack(wz_w3, OWZ3F, HID, HID, HID, 0);
    pack(wx_w0, OWX0F, DD,  HID, DD,  0);
    pack(wx_w1, OWX1F, DD,  HID, DD,  0);
    pack(wx_w2, OWX2F, DD,  HID, DD,  0);
    // backward orientation (K = output dim, N = input dim, element = Wsq[k][n])
    pack(wz_w3, OWZ3B, HID, HID, HID, 1);
    pack(wz_w2, OWZ2B, HID, HID, HID, 1);
    pack(wz_w1, OWZ1B, HID, HID, HID, 1);
    pack(wz_w0, OWZ0B, HID, DD,  DD,  1);
    pack(wx_w2, OWX2B, HID, DD,  DD,  1);
    pack(wx_w1, OWX1B, HID, DD,  DD,  1);
    pack(wx_w0, OWX0B, HID, DD,  DD,  1);

    icnn_sq_vec<<<2, 256, 0, stream>>>(wz_w4, wx_w3, fvec);

    icnn_main<<<BB / ROWS, 256, LDS_BYTES, stream>>>(
        state, wz_b0, wx_b0, wx_b1, wx_b2, wsh, fvec, (float*)d_out);
}
